// ARNN_5196910428286
// MI455X (gfx1250) — matrix operand order repaired and checked by reading, not compiled, not run on hardware
//
#include <hip/hip_runtime.h>
#include <math.h>

// ---------------- problem constants ----------------
#define T_STEPS 128
#define B_ROWS  128
#define D_IN    512
#define H_DIM   1024
#define O_DIM   512
#define KCAT    (D_IN + H_DIM)   // 1536
#define MAXP    6
#define NBLK    64
#define NTHR    256
#define NT_ALL  (NBLK * NTHR)

// padded LDS row strides (in shorts): stride*2 bytes == 4 DWORDs mod 64 banks -> conflict-free b128
#define W1_STRIDE (KCAT + 8)     // 1544
#define WO_STRIDE (H_DIM + 8)    // 1032
#define XB_STRIDE (D_IN + 8)     // 520

#define LDS_W1_OFF 0
#define LDS_WO_OFF (16 * W1_STRIDE)                    // 24704 shorts
#define LDS_XB_OFF (LDS_WO_OFF + 16 * WO_STRIDE)       // 41216 shorts
#define LDS_SHORTS (LDS_XB_OFF + B_ROWS * XB_STRIDE)   // 107776 shorts
#define LDS_BYTES  (LDS_SHORTS * 2)                    // 215552 bytes (<320KB WGP LDS)

typedef __attribute__((ext_vector_type(16))) __bf16 bf16x16;
typedef __attribute__((ext_vector_type(8)))  float  f32x8;
typedef __attribute__((ext_vector_type(4)))  unsigned int u32x4;
typedef __attribute__((ext_vector_type(4)))  unsigned int v4u;
typedef __attribute__((ext_vector_type(8)))  int v8i;
typedef __attribute__((ext_vector_type(4)))  int v4i;

union V16 { u32x4 q[2]; bf16x16 v; };

// ---------------- workspace layout (bytes) ----------------
constexpr size_t O_BIAS = 0;                                    // f32  [H] (bih+bhh)
constexpr size_t O_WIH0 = O_BIAS + (size_t)H_DIM * 4;           // f32  [H] (flag column)
constexpr size_t O_XB   = O_WIH0 + (size_t)H_DIM * 4;           // bf16 [B][D]
constexpr size_t O_SB0  = O_XB   + (size_t)B_ROWS * D_IN * 2;   // bf16 [B][H]
constexpr size_t O_SB1  = O_SB0  + (size_t)B_ROWS * H_DIM * 2;  // bf16 [B][H]
constexpr size_t O_SNF0 = O_SB1  + (size_t)B_ROWS * H_DIM * 2;  // f32  [B][H]
constexpr size_t O_SNF1 = O_SNF0 + (size_t)B_ROWS * H_DIM * 4;  // f32  [B][H]
constexpr size_t O_SACC = O_SNF1 + (size_t)B_ROWS * H_DIM * 4;  // f32  [B][H]
constexpr size_t O_PPART= O_SACC + (size_t)B_ROWS * H_DIM * 4;  // f32  [NBLK][B]
constexpr size_t O_PEMIT= O_PPART+ (size_t)NBLK * B_ROWS * 4;   // f32  [B] (padded 512)
constexpr size_t O_HALT = O_PEMIT+ 512;                         // f32  [B]
constexpr size_t O_HSUM = O_HALT + 512;                         // f32  [B]
constexpr size_t O_PSUM = O_HSUM + 512;                         // f32  [B]
constexpr size_t O_BAR  = O_PSUM + 512;                         // barrier (2 u32)

struct Bar { unsigned cnt; unsigned gen; };

__device__ __forceinline__ unsigned short f2bf(float f) {
    unsigned u = __float_as_uint(f);
    unsigned r = u + 0x7FFFu + ((u >> 16) & 1u);   // round-to-nearest-even
    return (unsigned short)(r >> 16);
}

// device-wide barrier (64 resident workgroups; generation counter)
__device__ __forceinline__ void gsync(Bar* bar) {
    __syncthreads();
    if (threadIdx.x == 0) {
        __threadfence();
        unsigned g = __hip_atomic_load(&bar->gen, __ATOMIC_ACQUIRE, __HIP_MEMORY_SCOPE_AGENT);
        unsigned a = __hip_atomic_fetch_add(&bar->cnt, 1u, __ATOMIC_ACQ_REL, __HIP_MEMORY_SCOPE_AGENT);
        if (a == (unsigned)(NBLK - 1)) {
            __hip_atomic_store(&bar->cnt, 0u, __ATOMIC_RELAXED, __HIP_MEMORY_SCOPE_AGENT);
            __hip_atomic_fetch_add(&bar->gen, 1u, __ATOMIC_RELEASE, __HIP_MEMORY_SCOPE_AGENT);
        } else {
            while (__hip_atomic_load(&bar->gen, __ATOMIC_ACQUIRE, __HIP_MEMORY_SCOPE_AGENT) == g)
                __builtin_amdgcn_s_sleep(1);
        }
    }
    __syncthreads();
}

// assemble a 16x32(bf16) A-fragment or 32x16(bf16) B-fragment from two 16B chunks
__device__ __forceinline__ bf16x16 load_frag(const unsigned short* p0, const unsigned short* p1) {
    V16 f;
    f.q[0] = *(const u32x4*)p0;
    f.q[1] = *(const u32x4*)p1;
    return f.v;
}

__global__ void bar_init_kernel(char* ws) {
    Bar* bar = (Bar*)(ws + O_BAR);
    bar->cnt = 0u;
    bar->gen = 0u;
}

__global__ __launch_bounds__(NTHR)
void arnn_persistent(const float* __restrict__ x,    // [B,T,D]
                     const float* __restrict__ s0,   // [1,B,H]
                     const float* __restrict__ Wih,  // [H, 1+D]
                     const float* __restrict__ bih,  // [H]
                     const float* __restrict__ Whh,  // [H,H]
                     const float* __restrict__ bhh,  // [H]
                     const float* __restrict__ Wh,   // [1,H]
                     const float* __restrict__ bhp,  // [1]
                     const float* __restrict__ Wo,   // [O,H]
                     const float* __restrict__ bo,   // [O]
                     float* __restrict__ out,        // [B,T,O] ++ pond [B,T]
                     char* __restrict__ ws)
{
    extern __shared__ unsigned short lds_all[];
    unsigned short* ldsW1 = lds_all + LDS_W1_OFF;   // [16][W1_STRIDE] this block's Wcat rows
    unsigned short* ldsWO = lds_all + LDS_WO_OFF;   // [16][WO_STRIDE] this block's Wo rows
    unsigned short* ldsXB = lds_all + LDS_XB_OFF;   // [128][XB_STRIDE] x_t tile (TDM-loaded)

    float* bias_s          = (float*)(ws + O_BIAS);
    float* wih0            = (float*)(ws + O_WIH0);
    unsigned short* xb     = (unsigned short*)(ws + O_XB);
    unsigned short* sbuf0  = (unsigned short*)(ws + O_SB0);
    unsigned short* sbuf1  = (unsigned short*)(ws + O_SB1);
    float* snf0            = (float*)(ws + O_SNF0);
    float* snf1            = (float*)(ws + O_SNF1);
    float* sacc            = (float*)(ws + O_SACC);
    float* ppart           = (float*)(ws + O_PPART);
    float* pemit           = (float*)(ws + O_PEMIT);
    float* halted          = (float*)(ws + O_HALT);
    float* hsum            = (float*)(ws + O_HSUM);
    float* psum            = (float*)(ws + O_PSUM);
    Bar* bar               = (Bar*)(ws + O_BAR);

    float* pond = out + (size_t)B_ROWS * T_STEPS * O_DIM;

    const int tid  = blockIdx.x * NTHR + threadIdx.x;
    const int wave = threadIdx.x >> 5;
    const int lane = threadIdx.x & 31;
    const int half = lane >> 4;
    const int ml   = lane & 15;

    // ---- one-time prep: stage this block's weight slices into LDS as bf16 ----
    {
        const int n0 = blockIdx.x << 4;
        for (int i = threadIdx.x; i < 16 * KCAT; i += NTHR) {
            int r = i / KCAT, k = i - r * KCAT;
            int n = n0 + r;
            float w = (k < D_IN) ? Wih[(size_t)n * (D_IN + 1) + 1 + k]
                                 : Whh[(size_t)n * H_DIM + (k - D_IN)];
            ldsW1[r * W1_STRIDE + k] = f2bf(w);
        }
        if (blockIdx.x < 32) {
            for (int i = threadIdx.x; i < 16 * H_DIM; i += NTHR) {
                int r = i / H_DIM, k = i - r * H_DIM;
                ldsWO[r * WO_STRIDE + k] = f2bf(Wo[(size_t)(n0 + r) * H_DIM + k]);
            }
        }
        for (int i = tid; i < H_DIM; i += NT_ALL) {
            bias_s[i] = bih[i] + bhh[i];
            wih0[i]   = Wih[(size_t)i * (D_IN + 1)];   // flag column
        }
    }

    for (int t = 0; t < T_STEPS; ++t) {
        gsync(bar);   // close prep / previous timestep's phase-3

        // ---- timestep prologue ----
        for (int i = tid; i < B_ROWS * D_IN; i += NT_ALL) {
            int b = i >> 9, d = i & (D_IN - 1);
            xb[i] = f2bf(x[((size_t)b * T_STEPS + t) * D_IN + d]);
        }
        for (int i = tid; i < B_ROWS * H_DIM; i += NT_ALL) {
            float sv = (t == 0) ? s0[i] : sacc[i];
            sbuf0[i] = f2bf(sv);
            sacc[i]  = 0.f;
        }
        for (int i = tid; i < B_ROWS * O_DIM; i += NT_ALL) {
            int b = i >> 9, n = i & (O_DIM - 1);
            out[((size_t)b * T_STEPS + t) * O_DIM + n] = 0.f;
        }
        if (tid < B_ROWS) {
            pond[tid * T_STEPS + t] = 0.f;
            halted[tid] = 0.f; hsum[tid] = 0.f; psum[tid] = 0.f;
        }
        gsync(bar);

        // ---- TDM: pull the bf16 x_t tile into LDS (padded rows), wave 0 issues ----
        if (threadIdx.x < 32) {
            unsigned long long ga = (unsigned long long)(uintptr_t)xb;
            unsigned ldsoff = (unsigned)(uintptr_t)ldsXB;   // low 32 bits = LDS byte offset
            v4u g0;
            g0[0] = 1u;                                          // count=1, user descriptor
            g0[1] = ldsoff;                                      // lds_addr
            g0[2] = (unsigned)(ga & 0xffffffffu);                // global_addr[31:0]
            g0[3] = (unsigned)((ga >> 32) & 0x01ffffffu)         // global_addr[56:32]
                    | (2u << 30);                                // type=2 (image)
            v8i g1;
            g1[0] = (int)((1u << 16)      // data_size=1 -> 2 bytes
                    | (1u << 20)          // pad_enable
                    | (7u << 22)          // pad_interval: 256 DWORDs (= one 512-elem row)
                    | (3u << 25));        // pad_amount: 4 DWORDs (= 8 shorts)
            g1[1] = (int)((unsigned)D_IN << 16);    // tensor_dim0[15:0] at bits[63:48]
            g1[2] = (int)((unsigned)B_ROWS << 16);  // tensor_dim0 hi=0 | tensor_dim1 lo
            g1[3] = (int)((unsigned)D_IN << 16);    // tensor_dim1 hi=0 | tile_dim0=512
            g1[4] = (int)B_ROWS;                    // tile_dim1=128, tile_dim2=0
            g1[5] = (int)D_IN;                      // tensor_dim0_stride = 512 elems
            g1[6] = 0;
            g1[7] = 0;
            v4i gz4 = {0, 0, 0, 0};
            v8i gz8 = {0, 0, 0, 0, 0, 0, 0, 0};
            __builtin_amdgcn_tensor_load_to_lds(g0, g1, gz4, gz4, gz8, 0);
            __builtin_amdgcn_s_wait_tensorcnt(0);
        }
        __syncthreads();

        int cur = 0;
        for (int np = 0; np < MAXP; ++np) {
            // ========== phase 1: s_new = tanh(A @ Wcat^T + bias) ==========
            {
                const int n0 = blockIdx.x << 4;     // 64 blocks cover H=1024
                const int m0 = wave << 4;           // 8 waves cover B=128
                const unsigned short* sprev = cur ? sbuf1 : sbuf0;
                unsigned short* scur_b      = cur ? sbuf0 : sbuf1;
                float* scur_f               = cur ? snf0  : snf1;

                f32x8 acc = {0.f, 0.f, 0.f, 0.f, 0.f, 0.f, 0.f, 0.f};
                const unsigned short* axrow = ldsXB + (size_t)(m0 + ml) * XB_STRIDE;  // LDS
                const unsigned short* asrow = sprev + (size_t)(m0 + ml) * H_DIM;      // global
                const unsigned short* brow  = ldsW1 + (size_t)ml * W1_STRIDE + half * 16; // LDS

                #pragma unroll 4
                for (int kc = 0; kc < D_IN / 32; ++kc) {          // x part: K 0..511
                    int k0 = kc * 32;
                    bf16x16 A  = load_frag(axrow + k0 + half * 16, axrow + k0 + half * 16 + 8);
                    bf16x16 Bm = load_frag(brow + k0, brow + k0 + 8);
                    acc = __builtin_amdgcn_wmma_f32_16x16x32_bf16(false, A, false, Bm,
                                                                  (short)0, acc, false, false);
                }
                #pragma unroll 4
                for (int kc = 0; kc < H_DIM / 32; ++kc) {         // s part: K 512..1535
                    int k0 = kc * 32;
                    bf16x16 A  = load_frag(asrow + k0 + half * 16, asrow + k0 + half * 16 + 8);
                    bf16x16 Bm = load_frag(brow + D_IN + k0, brow + D_IN + k0 + 8);
                    acc = __builtin_amdgcn_wmma_f32_16x16x32_bf16(false, A, false, Bm,
                                                                  (short)0, acc, false, false);
                }

                const int n   = n0 + ml;
                const float bs  = bias_s[n] + ((np == 0) ? wih0[n] : 0.f);
                const float whn = Wh[n];
                #pragma unroll
                for (int r = 0; r < 8; ++r) {
                    int m = m0 + r + (half << 3);
                    float s = tanhf(acc[r] + bs);
                    scur_f[(size_t)m * H_DIM + n] = s;
                    scur_b[(size_t)m * H_DIM + n] = f2bf(s);
                    // fused halt-GEMV partial: reduce over the block's 16 columns
                    float p = whn * s;
                    p += __shfl_xor(p, 1, 32);
                    p += __shfl_xor(p, 2, 32);
                    p += __shfl_xor(p, 4, 32);
                    p += __shfl_xor(p, 8, 32);
                    if (ml == 0) ppart[blockIdx.x * B_ROWS + m] = p;   // single writer
                }
            }
            cur ^= 1;
            gsync(bar);

            // ========== phase 2: ACT halting update (block 0, 128 rows) ==========
            if (blockIdx.x == 0 && threadIdx.x < B_ROWS) {
                int b = threadIdx.x;
                float logit = bhp[0];
                for (int blk = 0; blk < NBLK; ++blk) logit += ppart[blk * B_ROWS + b];
                float p_raw = 1.f / (1.f + expf(-logit));
                float h_ = halted[b], hs = hsum[b], ps = psum[b];
                float cum = hs + p_raw;
                float tmp = ((1.f - h_) * cum >= 0.99f) ? 1.f : 0.f;
                if (np == MAXP - 1) tmp = 1.f - h_;
                float rr = tmp * (1.f - ps);
                float pe = (p_raw * (1.f - tmp) + rr) * (1.f - h_);
                pond[b * T_STEPS + t] += (rr + (float)(np + 1)) * tmp;
                hsum[b]   = hs + pe;
                psum[b]   = ps + pe;
                halted[b] = h_ + tmp;
                pemit[b]  = pe;
            }
            gsync(bar);

            // ========== phase 3: out GEMM + weighted accumulation ==========
            const unsigned short* snb = cur ? sbuf1 : sbuf0;
            const float*          snf = cur ? snf1  : snf0;
            if (blockIdx.x < 32) {
                // out = s_new @ Wo^T + bo ; out_acc += p_emit * out (direct to d_out)
                const int n0 = blockIdx.x << 4;     // 32 blocks cover O=512
                const int m0 = wave << 4;
                f32x8 acc = {0.f, 0.f, 0.f, 0.f, 0.f, 0.f, 0.f, 0.f};
                const unsigned short* arow = snb + (size_t)(m0 + ml) * H_DIM;        // global
                const unsigned short* brow = ldsWO + (size_t)ml * WO_STRIDE + half * 16; // LDS
                #pragma unroll 4
                for (int kc = 0; kc < H_DIM / 32; ++kc) {
                    int k0 = kc * 32;
                    bf16x16 A  = load_frag(arow + k0 + half * 16, arow + k0 + half * 16 + 8);
                    bf16x16 Bm = load_frag(brow + k0, brow + k0 + 8);
                    acc = __builtin_amdgcn_wmma_f32_16x16x32_bf16(false, A, false, Bm,
                                                                  (short)0, acc, false, false);
                }
                const int n = n0 + ml;
                const float bon = bo[n];
                #pragma unroll
                for (int r = 0; r < 8; ++r) {
                    int m = m0 + r + (half << 3);
                    float o = acc[r] + bon;
                    size_t oi = ((size_t)m * T_STEPS + t) * O_DIM + n;
                    out[oi] += pemit[m] * o;          // unique owner per element
                }
            } else {
                // s_acc += p_emit * s_new (elementwise, 256 waves x 512 elems)
                int w2 = (blockIdx.x - 32) * 8 + wave;
                int base = w2 * 512;
                #pragma unroll
                for (int i2 = 0; i2 < 16; ++i2) {
                    int idx = base + i2 * 32 + lane;
                    int b = idx >> 10;
                    sacc[idx] += pemit[b] * snf[idx];
                }
            }
            // next phase-1 follows with no gsync; it writes the other sbuf/snf pair
        }
    }
}

extern "C" void kernel_launch(void* const* d_in, const int* in_sizes, int n_in,
                              void* d_out, int out_size, void* d_ws, size_t ws_size,
                              hipStream_t stream) {
    const float* x   = (const float*)d_in[0];
    const float* s0  = (const float*)d_in[1];
    const float* Wih = (const float*)d_in[2];
    const float* bih = (const float*)d_in[3];
    const float* Whh = (const float*)d_in[4];
    const float* bhh = (const float*)d_in[5];
    const float* Wh  = (const float*)d_in[6];
    const float* bhp = (const float*)d_in[7];
    const float* Wo  = (const float*)d_in[8];
    const float* bo  = (const float*)d_in[9];

    bar_init_kernel<<<1, 1, 0, stream>>>((char*)d_ws);
    arnn_persistent<<<NBLK, NTHR, LDS_BYTES, stream>>>(x, s0, Wih, bih, Whh, bhh, Wh, bhp,
                                                       Wo, bo, (float*)d_out, (char*)d_ws);
}
